// SSD_59021440581846
// MI455X (gfx1250) — matrix operand order repaired and checked by reading, not compiled, not run on hardware
//
#include <hip/hip_runtime.h>

// ---------------------------------------------------------------------------
// CDNA5 (gfx1250) SSD300 forward.
// Implicit-GEMM conv on V_WMMA_F32_16X16X32_BF16, NHWC bf16 activations,
// pre-packed bf16 weights in A-fragment layout.
// Workgroup = 8 waves sharing a double-buffered LDS B-tile (64 px x 32 ch);
// 1x1 convs stage the B-tile with the Tensor Data Mover when available.
// ---------------------------------------------------------------------------

typedef __attribute__((ext_vector_type(16))) __bf16 v16bf;
typedef __attribute__((ext_vector_type(8)))  __bf16 v8bf;
typedef __attribute__((ext_vector_type(8)))  float  v8f;

#if __has_builtin(__builtin_amdgcn_tensor_load_to_lds)
#define SSD_HAVE_TDM 1
#else
#define SSD_HAVE_TDM 0
#endif

__device__ __forceinline__ __bf16 f2bf(float f) {
    unsigned u = __builtin_bit_cast(unsigned, f);
    unsigned r = (u + 0x7FFFu + ((u >> 16) & 1u)) >> 16;   // RNE
    unsigned short s = (unsigned short)r;
    return __builtin_bit_cast(__bf16, s);
}
__device__ __forceinline__ float bf2f(__bf16 b) {
    unsigned u = (unsigned)__builtin_bit_cast(unsigned short, b) << 16;
    return __builtin_bit_cast(float, u);
}

__device__ __forceinline__ v8f wmma_bf16(v16bf a, v16bf b, v8f c) {
    return __builtin_amdgcn_wmma_f32_16x16x32_bf16(false, a, false, b, (short)0, c, false, false);
}

// ---------------------------------------------------------------------------
// Weight pack: OIHW fp32 -> bf16 fragment layout [mt][kk][cb][lane][16].
// A-frag lane K map: half 0 -> K {0..15}, half 1 -> K {16..31}.
// ---------------------------------------------------------------------------
__global__ void ssd_pack_w(const float* __restrict__ w, __bf16* __restrict__ wp,
                           int Co, int CiSrc, int cbs, int KK, long total)
{
    long idx = (long)blockIdx.x * blockDim.x + threadIdx.x;
    if (idx >= total) return;
    int i    = (int)(idx & 15);
    int lane = (int)((idx >> 4) & 31);
    long t   = idx >> 9;
    int cb   = (int)(t % cbs); t /= cbs;
    int kk   = (int)(t % KK);
    int mt   = (int)(t / KK);
    int m    = mt * 16 + (lane & 15);
    int kbA  = (lane >> 4) * 16;
    int ci   = cb * 32 + kbA + i;
    float v  = 0.f;
    if (m < Co && ci < CiSrc)
        v = w[((long)m * CiSrc + ci) * KK + kk];
    wp[idx] = f2bf(v);
}

// ---------------------------------------------------------------------------
// Generic conv (3x3 / strided / dilated / heads), LDS-staged B tile.
// Block: 256 thr = 8 waves. Macro tile: WMT*16 (Co) x 64 (pixels).
// Double-buffered 4KB B stages; one barrier per K-stage.
// ---------------------------------------------------------------------------
template <typename OUT_T>
__global__ __launch_bounds__(256)
void ssd_conv_wmma(const __bf16* __restrict__ x, const __bf16* __restrict__ wp,
                   const float* __restrict__ bias, OUT_T* __restrict__ y,
                   int N, int Ci, int Hi, int Wi,
                   int Co, int Ho, int Wo,
                   int KH, int KW, int stride, int pad, int dil, int relu,
                   long imgStrideOut, long pixBaseOut)
{
    __shared__ __bf16 lds[2][64 * 32];

    const int tid  = threadIdx.x;
    const int lane = tid & 31;
    const int wave = tid >> 5;
    const int half = lane >> 4;
    const int l15  = lane & 15;

    const int P      = N * Ho * Wo;
    const int mtiles = (Co + 15) >> 4;
    const int WMT    = mtiles < 8 ? mtiles : 8;
    const int n64    = (P + 63) >> 6;
    const int mg     = blockIdx.x / n64;
    const int nt     = blockIdx.x % n64;
    const int mt     = mg * WMT + wave;
    const bool computeActive = (wave < WMT) && (mt < mtiles);

    const int cbs   = Ci >> 5;
    const int KK    = KH * KW;
    const int total = KK * cbs;

    // staging role: thread -> (pixel 0..63, 8-channel chunk)
    const int tpix = tid >> 2;
    const int ch8  = (tid & 3) * 8;
    int sImg, sOh, sOw;
    {
        int n = nt * 64 + tpix;
        int p = (n < P) ? n : (P - 1);
        sOw = p % Wo; int t = p / Wo; sOh = t % Ho; sImg = t / Ho;
    }

    // compute role: 4 pixels per lane
    int img[4], oh[4], ow[4];
    #pragma unroll
    for (int j = 0; j < 4; ++j) {
        int n = nt * 64 + j * 16 + l15;
        int p = (n < P) ? n : (P - 1);
        ow[j] = p % Wo; int t = p / Wo; oh[j] = t % Ho; img[j] = t / Ho;
    }

    v8f acc0, acc1, acc2, acc3;
    #pragma unroll
    for (int v = 0; v < 8; ++v) { acc0[v]=0.f; acc1[v]=0.f; acc2[v]=0.f; acc3[v]=0.f; }

    // packed-A stream is linear in the stage index s
    const __bf16* wk = wp + (long)mt * total * 512 + (long)lane * 16;

    int cb_n = 0, kh_n = 0, kw_n = 0;            // counters of next stage to issue
    auto stage = [&](int buf) {
        int ih = sOh * stride - pad + kh_n * dil;
        int iw = sOw * stride - pad + kw_n * dil;
        v8bf c;
        #pragma unroll
        for (int i = 0; i < 8; ++i) c[i] = f2bf(0.f);
        if (ih >= 0 && ih < Hi && iw >= 0 && iw < Wi) {
            const __bf16* src = x + (((long)sImg * Hi + ih) * Wi + iw) * Ci + cb_n * 32 + ch8;
            c = *(const v8bf*)src;
            __builtin_prefetch((const void*)(src + 32), 0, 1);  // next channel block
        }
        *(v8bf*)(&lds[buf][tpix * 32 + ch8]) = c;
        if (++cb_n == cbs) { cb_n = 0; if (++kw_n == KW) { kw_n = 0; ++kh_n; } }
    };

    stage(0);
    for (int s = 0; s < total; ++s) {
        __syncthreads();
        if (s + 1 < total) stage((s + 1) & 1);
        if (computeActive) {
            v16bf a = *(const v16bf*)(wk + (long)s * 512);
            const __bf16* lb = &lds[s & 1][0];
            v16bf b0 = *(const v16bf*)(lb + (0 * 16 + l15) * 32 + half * 16);
            v16bf b1 = *(const v16bf*)(lb + (1 * 16 + l15) * 32 + half * 16);
            v16bf b2 = *(const v16bf*)(lb + (2 * 16 + l15) * 32 + half * 16);
            v16bf b3 = *(const v16bf*)(lb + (3 * 16 + l15) * 32 + half * 16);
            acc0 = wmma_bf16(a, b0, acc0);
            acc1 = wmma_bf16(a, b1, acc1);
            acc2 = wmma_bf16(a, b2, acc2);
            acc3 = wmma_bf16(a, b3, acc3);
        }
    }
    if (!computeActive) return;

    const int chBase = mt * 16 + 8 * half;       // C/D: VGPR v -> ch chBase+v
    float bi[8];
    #pragma unroll
    for (int v = 0; v < 8; ++v) bi[v] = (chBase + v < Co) ? bias[chBase + v] : 0.f;

    v8f accs[4] = { acc0, acc1, acc2, acc3 };
    #pragma unroll
    for (int j = 0; j < 4; ++j) {
        int nn = nt * 64 + j * 16 + l15;
        if (nn >= P) continue;
        long dst = (long)img[j] * imgStrideOut + pixBaseOut
                 + ((long)oh[j] * Wo + ow[j]) * Co + chBase;
        if constexpr (sizeof(OUT_T) == 2) {
            if (chBase + 8 <= Co) {
                v8bf o;
                #pragma unroll
                for (int v = 0; v < 8; ++v) {
                    float r = accs[j][v] + bi[v];
                    if (relu) r = fmaxf(r, 0.f);
                    o[v] = f2bf(r);
                }
                *(v8bf*)(y + dst) = o;
            } else {
                #pragma unroll
                for (int v = 0; v < 8; ++v)
                    if (chBase + v < Co) {
                        float r = accs[j][v] + bi[v];
                        if (relu) r = fmaxf(r, 0.f);
                        y[dst + v] = (OUT_T)f2bf(r);
                    }
            }
        } else {
            #pragma unroll
            for (int v = 0; v < 8; ++v)
                if (chBase + v < Co) {
                    float r = accs[j][v] + bi[v];
                    if (relu) r = fmaxf(r, 0.f);
                    y[dst + v] = r;
                }
        }
    }
}

// ---------------------------------------------------------------------------
// 1x1 conv = GEMM. B tile (64 consecutive NHWC pixel rows x 32 ch) staged by
// the Tensor Data Mover (wave 0 issues, s_wait_tensorcnt, barrier publishes),
// double-buffered; falls back to cooperative ds staging without TDM builtin.
// ---------------------------------------------------------------------------
__device__ __forceinline__ void ssd_tdm_issue(unsigned ldsByteAddr, const __bf16* gsrc,
                                              int dim0Rem, int dim1Rem, int strideElems)
{
#if SSD_HAVE_TDM
    typedef unsigned u32x4 __attribute__((ext_vector_type(4)));
    typedef int      i32x4 __attribute__((ext_vector_type(4)));
    typedef int      i32x8 __attribute__((ext_vector_type(8)));
    unsigned long long ga = (unsigned long long)gsrc;
    u32x4 g0;
    g0[0] = 1u;                                            // count=1, user D#
    g0[1] = ldsByteAddr;                                   // lds_addr
    g0[2] = (unsigned)(ga & 0xFFFFFFFFu);                  // global_addr[31:0]
    g0[3] = (unsigned)((ga >> 32) & 0x01FFFFFFu) | (2u << 30); // addr[56:32], type=2
    i32x8 g1;
    g1[0] = 0x00010000;                                    // data_size=1 (2B), no flags
    g1[1] = (int)(((unsigned)dim0Rem & 0xFFFFu) << 16);    // tensor_dim0[15:0]
    g1[2] = (int)((((unsigned)dim0Rem >> 16) & 0xFFFFu)
                | (((unsigned)dim1Rem & 0xFFFFu) << 16));  // dim0 hi | dim1 lo
    g1[3] = (int)((((unsigned)dim1Rem >> 16) & 0xFFFFu)
                | (32u << 16));                            // dim1 hi | tile_dim0=32
    g1[4] = 64;                                            // tile_dim1=64, tile_dim2=0
    g1[5] = strideElems;                                   // tensor_dim0_stride[31:0]
    g1[6] = 0;
    g1[7] = 0;
    i32x4 z4; z4[0] = 0; z4[1] = 0; z4[2] = 0; z4[3] = 0;  // groups 2/3 unused (2D)
    i32x8 z8;
    #pragma unroll
    for (int i = 0; i < 8; ++i) z8[i] = 0;                 // extra group (6-arg form)
    __builtin_amdgcn_tensor_load_to_lds(g0, g1, z4, z4, z8, 0);
#else
    (void)ldsByteAddr; (void)gsrc; (void)dim0Rem; (void)dim1Rem; (void)strideElems;
#endif
}

__global__ __launch_bounds__(256)
void ssd_conv1x1_wmma(const __bf16* __restrict__ x, const __bf16* __restrict__ wp,
                      const float* __restrict__ bias, __bf16* __restrict__ y,
                      int P, int Ci, int Co)
{
    __shared__ __bf16 lds[2][64 * 32];

    const int tid  = threadIdx.x;
    const int lane = tid & 31;
    const int wave = tid >> 5;
    const int half = lane >> 4;
    const int l15  = lane & 15;

    const int mtiles = (Co + 15) >> 4;
    const int WMT    = mtiles < 8 ? mtiles : 8;
    const int n64    = (P + 63) >> 6;
    const int mg     = blockIdx.x / n64;
    const int nt     = blockIdx.x % n64;
    const int mt     = mg * WMT + wave;
    const bool computeActive = (wave < WMT) && (mt < mtiles);

    const int cbs = Ci >> 5;
    const int dim1Rem = P - nt * 64;                 // tail rows -> TDM OOB zeros
    const __bf16* gbase = x + (long)nt * 64 * Ci;

    v8f acc0, acc1, acc2, acc3;
    #pragma unroll
    for (int v = 0; v < 8; ++v) { acc0[v]=0.f; acc1[v]=0.f; acc2[v]=0.f; acc3[v]=0.f; }

    const __bf16* wk = wp + (long)mt * cbs * 512 + (long)lane * 16;

#if SSD_HAVE_TDM
    if (wave == 0)
        ssd_tdm_issue((unsigned)(unsigned long long)&lds[0][0], gbase, Ci, dim1Rem, Ci);
#else
    {   // cooperative fallback stage 0
        const int tpix = tid >> 2, ch8 = (tid & 3) * 8;
        v8bf c;
        #pragma unroll
        for (int i = 0; i < 8; ++i) c[i] = f2bf(0.f);
        if (nt * 64 + tpix < P) c = *(const v8bf*)(gbase + (long)tpix * Ci + ch8);
        *(v8bf*)(&lds[0][tpix * 32 + ch8]) = c;
    }
#endif

    for (int s = 0; s < cbs; ++s) {
#if SSD_HAVE_TDM
        if (wave == 0) __builtin_amdgcn_s_wait_tensorcnt((short)0);
#endif
        __syncthreads();
        if (s + 1 < cbs) {
#if SSD_HAVE_TDM
            if (wave == 0)
                ssd_tdm_issue((unsigned)(unsigned long long)&lds[(s + 1) & 1][0],
                              gbase + (s + 1) * 32, Ci - (s + 1) * 32, dim1Rem, Ci);
#else
            const int tpix = tid >> 2, ch8 = (tid & 3) * 8;
            v8bf c;
            #pragma unroll
            for (int i = 0; i < 8; ++i) c[i] = f2bf(0.f);
            if (nt * 64 + tpix < P)
                c = *(const v8bf*)(gbase + (long)tpix * Ci + (s + 1) * 32 + ch8);
            *(v8bf*)(&lds[(s + 1) & 1][tpix * 32 + ch8]) = c;
#endif
        }
        if (computeActive) {
            v16bf a = *(const v16bf*)(wk + (long)s * 512);
            const __bf16* lb = &lds[s & 1][0];
            v16bf b0 = *(const v16bf*)(lb + (0 * 16 + l15) * 32 + half * 16);
            v16bf b1 = *(const v16bf*)(lb + (1 * 16 + l15) * 32 + half * 16);
            v16bf b2 = *(const v16bf*)(lb + (2 * 16 + l15) * 32 + half * 16);
            v16bf b3 = *(const v16bf*)(lb + (3 * 16 + l15) * 32 + half * 16);
            acc0 = wmma_bf16(a, b0, acc0);
            acc1 = wmma_bf16(a, b1, acc1);
            acc2 = wmma_bf16(a, b2, acc2);
            acc3 = wmma_bf16(a, b3, acc3);
        }
    }
    if (!computeActive) return;

    const int chBase = mt * 16 + 8 * half;
    float bi[8];
    #pragma unroll
    for (int v = 0; v < 8; ++v) bi[v] = (chBase + v < Co) ? bias[chBase + v] : 0.f;

    v8f accs[4] = { acc0, acc1, acc2, acc3 };
    #pragma unroll
    for (int j = 0; j < 4; ++j) {
        int nn = nt * 64 + j * 16 + l15;
        if (nn >= P) continue;
        long dst = (long)nn * Co + chBase;
        if (chBase + 8 <= Co) {
            v8bf o;
            #pragma unroll
            for (int v = 0; v < 8; ++v) o[v] = f2bf(fmaxf(accs[j][v] + bi[v], 0.f));
            *(v8bf*)(y + dst) = o;                 // 1x1 convs are all ReLU'd
        } else {
            #pragma unroll
            for (int v = 0; v < 8; ++v)
                if (chBase + v < Co) y[dst + v] = f2bf(fmaxf(accs[j][v] + bi[v], 0.f));
        }
    }
}

// NCHW fp32 -> NHWC bf16 with channel zero-padding to Cpad
__global__ void ssd_nchw_to_nhwc(const float* __restrict__ x, __bf16* __restrict__ y,
                                 int N, int C, int H, int W, int Cpad)
{
    long idx   = (long)blockIdx.x * blockDim.x + threadIdx.x;
    long total = (long)N * H * W * Cpad;
    if (idx >= total) return;
    int c = (int)(idx % Cpad); long t = idx / Cpad;
    int w = (int)(t % W); t /= W;
    int h = (int)(t % H);
    int n = (int)(t / H);
    float v = (c < C) ? x[(((long)n * C + c) * H + h) * W + w] : 0.f;
    y[idx] = f2bf(v);
}

// NHWC bf16 maxpool (low-side padding only; ceil handled via host Ho/Wo)
__global__ void ssd_maxpool_nhwc(const __bf16* __restrict__ x, __bf16* __restrict__ y,
                                 int N, int C, int Hi, int Wi, int Ho, int Wo,
                                 int win, int stride, int padLo)
{
    long idx   = (long)blockIdx.x * blockDim.x + threadIdx.x;
    long total = (long)N * Ho * Wo * C;
    if (idx >= total) return;
    int c  = (int)(idx % C); long t = idx / C;
    int ow = (int)(t % Wo);  t /= Wo;
    int oh = (int)(t % Ho);
    int n  = (int)(t / Ho);
    float m = -__builtin_huge_valf();
    int h0 = oh * stride - padLo, w0 = ow * stride - padLo;
    for (int i = 0; i < win; ++i) {
        int ih = h0 + i; if (ih < 0 || ih >= Hi) continue;
        for (int j = 0; j < win; ++j) {
            int iw = w0 + j; if (iw < 0 || iw >= Wi) continue;
            m = fmaxf(m, bf2f(x[(((long)n * Hi + ih) * Wi + iw) * C + c]));
        }
    }
    y[idx] = f2bf(m);
}

// In-place channel L2-norm + learned rescale (conv4_3), NHWC bf16
__global__ void ssd_l2norm_nhwc(__bf16* __restrict__ x, const float* __restrict__ rescale,
                                long NP, int C)
{
    long idx = (long)blockIdx.x * blockDim.x + threadIdx.x;
    if (idx >= NP) return;
    __bf16* p = x + idx * C;
    float s = 0.f;
    for (int c = 0; c < C; ++c) { float v = bf2f(p[c]); s += v * v; }
    float inv = rsqrtf(s);
    for (int c = 0; c < C; ++c) p[c] = f2bf(bf2f(p[c]) * inv * rescale[c]);
}

// ---------------------------------------------------------------------------

extern "C" void kernel_launch(void* const* d_in, const int* in_sizes, int n_in,
                              void* d_out, int out_size, void* d_ws, size_t ws_size,
                              hipStream_t stream)
{
    (void)in_sizes; (void)n_in; (void)out_size; (void)ws_size;

    const float* image = (const float*)d_in[0];
    auto W_ = [&](int i) { return (const float*)d_in[1 + 2 * i]; };
    auto B_ = [&](int i) { return (const float*)d_in[2 + 2 * i]; };
    const float* rescale = (const float*)d_in[71];

    // ---- workspace carve (bf16; all sizes multiples of 16 elems) ----
    __bf16* imgT = (__bf16*)d_ws;              // 16*300*300*32 = 46,080,000
    __bf16* bufA = imgT + 46080000L;           // 16*300*300*64 = 92,160,000
    __bf16* bufB = bufA + 92160000L;
    __bf16* f4   = bufB + 92160000L;           // 16*38*38*512  = 11,829,248
    __bf16* f7   = f4   + 11829248L;           // 16*19*19*1024 =  5,914,624
    __bf16* f8   = f7   + 5914624L;            // 16*10*10*512  =    819,200
    __bf16* f9   = f8   + 819200L;             // 16*5*5*256    =    102,400
    __bf16* f10  = f9   + 102400L;             // 16*3*3*256    =     36,864
    __bf16* f11  = f10  + 36864L;              // 16*1*1*256    =      4,096
    __bf16* wpackCur = f11 + 4096L;            // packed weights (~30M bf16)

    const int N = 16;
    const int TB = 8732;

    auto grids = [&](int Co, long P) {
        int mtiles = (Co + 15) / 16;
        int WMT = mtiles < 8 ? mtiles : 8;
        long groups = (mtiles + WMT - 1) / WMT;
        long n64 = (P + 63) / 64;
        return (unsigned)(groups * n64);
    };

    // pack + conv launcher. outKind: 0 = bf16 NHWC intermediate, 1 = fp32 head
    auto conv = [&](const __bf16* x, int pi, void* y, int outKind,
                    int CiSrc, int CiPad, int Hi, int Wi, int Co, int Ho, int Wo,
                    int K, int s, int p, int d, int relu,
                    long imgStrideOut, long pixBaseOut) {
        const int KK = K * K;
        const int cbs = CiPad >> 5;
        const int mtiles = (Co + 15) >> 4;
        long psz = (long)mtiles * KK * cbs * 512;
        __bf16* wp = wpackCur; wpackCur += psz;
        ssd_pack_w<<<(unsigned)((psz + 255) / 256), 256, 0, stream>>>(
            W_(pi), wp, Co, CiSrc, cbs, KK, psz);
        long P = (long)N * Ho * Wo;
        unsigned grid = grids(Co, P);
        if (outKind == 0 && K == 1 && s == 1 && p == 0) {
            ssd_conv1x1_wmma<<<grid, 256, 0, stream>>>(
                x, wp, B_(pi), (__bf16*)y, (int)P, CiPad, Co);
        } else if (outKind == 0) {
            ssd_conv_wmma<__bf16><<<grid, 256, 0, stream>>>(
                x, wp, B_(pi), (__bf16*)y, N, CiPad, Hi, Wi, Co, Ho, Wo,
                K, K, s, p, d, relu, imgStrideOut, pixBaseOut);
        } else {
            ssd_conv_wmma<float><<<grid, 256, 0, stream>>>(
                x, wp, B_(pi), (float*)y, N, CiPad, Hi, Wi, Co, Ho, Wo,
                K, K, s, p, d, relu, imgStrideOut, pixBaseOut);
        }
    };
    auto pool = [&](const __bf16* x, __bf16* y, int C, int Hi, int Wi,
                    int Ho, int Wo, int win, int s, int padLo) {
        long tot = (long)N * Ho * Wo * C;
        ssd_maxpool_nhwc<<<(unsigned)((tot + 255) / 256), 256, 0, stream>>>(
            x, y, N, C, Hi, Wi, Ho, Wo, win, s, padLo);
    };
    auto istride = [&](int Ho, int Wo, int Co) { return (long)Ho * Wo * Co; };

    // ---- image: NCHW fp32 -> NHWC bf16 padded to 32 ch ----
    {
        long tot = (long)N * 300 * 300 * 32;
        ssd_nchw_to_nhwc<<<(unsigned)((tot + 255) / 256), 256, 0, stream>>>(
            image, imgT, N, 3, 300, 300, 32);
    }

    // ---- VGG base ----
    conv(imgT, 0, bufA, 0,   3,  32, 300, 300,  64, 300, 300, 3, 1, 1, 1, 1, istride(300,300,64), 0);
    conv(bufA, 1, bufB, 0,  64,  64, 300, 300,  64, 300, 300, 3, 1, 1, 1, 1, istride(300,300,64), 0);
    pool(bufB, bufA,  64, 300, 300, 150, 150, 2, 2, 0);
    conv(bufA, 2, bufB, 0,  64,  64, 150, 150, 128, 150, 150, 3, 1, 1, 1, 1, istride(150,150,128), 0);
    conv(bufB, 3, bufA, 0, 128, 128, 150, 150, 128, 150, 150, 3, 1, 1, 1, 1, istride(150,150,128), 0);
    pool(bufA, bufB, 128, 150, 150,  75,  75, 2, 2, 0);
    conv(bufB, 4, bufA, 0, 128, 128,  75,  75, 256,  75,  75, 3, 1, 1, 1, 1, istride(75,75,256), 0);
    conv(bufA, 5, bufB, 0, 256, 256,  75,  75, 256,  75,  75, 3, 1, 1, 1, 1, istride(75,75,256), 0);
    conv(bufB, 6, bufA, 0, 256, 256,  75,  75, 256,  75,  75, 3, 1, 1, 1, 1, istride(75,75,256), 0);
    pool(bufA, bufB, 256,  75,  75,  38,  38, 2, 2, 0);           // ceil via Ho=38
    conv(bufB, 7, bufA, 0, 256, 256,  38,  38, 512,  38,  38, 3, 1, 1, 1, 1, istride(38,38,512), 0);
    conv(bufA, 8, bufB, 0, 512, 512,  38,  38, 512,  38,  38, 3, 1, 1, 1, 1, istride(38,38,512), 0);
    conv(bufB, 9, f4,   0, 512, 512,  38,  38, 512,  38,  38, 3, 1, 1, 1, 1, istride(38,38,512), 0);
    pool(f4, bufA, 512,  38,  38,  19,  19, 2, 2, 0);
    conv(bufA, 10, bufB, 0, 512, 512, 19, 19, 512, 19, 19, 3, 1, 1, 1, 1, istride(19,19,512), 0);
    conv(bufB, 11, bufA, 0, 512, 512, 19, 19, 512, 19, 19, 3, 1, 1, 1, 1, istride(19,19,512), 0);
    conv(bufA, 12, bufB, 0, 512, 512, 19, 19, 512, 19, 19, 3, 1, 1, 1, 1, istride(19,19,512), 0);
    pool(bufB, bufA, 512, 19, 19, 19, 19, 3, 1, 1);               // k3 s1 p1
    conv(bufA, 13, bufB, 0,  512,  512, 19, 19, 1024, 19, 19, 3, 1, 6, 6, 1, istride(19,19,1024), 0); // atrous
    conv(bufB, 14, f7,   0, 1024, 1024, 19, 19, 1024, 19, 19, 1, 1, 0, 1, 1, istride(19,19,1024), 0); // TDM 1x1

    // ---- L2-normalize + rescale conv4_3 (in place, NHWC) ----
    {
        long np = (long)N * 38 * 38;
        ssd_l2norm_nhwc<<<(unsigned)((np + 255) / 256), 256, 0, stream>>>(
            f4, rescale, np, 512);
    }

    // ---- Auxiliary convolutions ----
    conv(f7,   15, bufA, 0, 1024, 1024, 19, 19, 256, 19, 19, 1, 1, 0, 1, 1, istride(19,19,256), 0); // TDM 1x1
    conv(bufA, 16, f8,   0,  256,  256, 19, 19, 512, 10, 10, 3, 2, 1, 1, 1, istride(10,10,512), 0);
    conv(f8,   17, bufA, 0,  512,  512, 10, 10, 128, 10, 10, 1, 1, 0, 1, 1, istride(10,10,128), 0); // TDM 1x1
    conv(bufA, 18, f9,   0,  128,  128, 10, 10, 256,  5,  5, 3, 2, 1, 1, 1, istride(5,5,256), 0);
    conv(f9,   19, bufA, 0,  256,  256,  5,  5, 128,  5,  5, 1, 1, 0, 1, 1, istride(5,5,128), 0);   // TDM 1x1
    conv(bufA, 20, f10,  0,  128,  128,  5,  5, 256,  3,  3, 3, 1, 0, 1, 1, istride(3,3,256), 0);
    conv(f10,  21, bufA, 0,  256,  256,  3,  3, 128,  3,  3, 1, 1, 0, 1, 1, istride(3,3,128), 0);   // TDM 1x1
    conv(bufA, 22, f11,  0,  128,  128,  3,  3, 256,  1,  1, 3, 1, 0, 1, 1, istride(1,1,256), 0);

    // ---- Prediction heads: NHWC fp32 written straight into d_out ----
    float* locs = (float*)d_out;
    float* scrs = locs + (long)N * TB * 4;
    struct Feat { const __bf16* x; int C, H, W, boxOff; };
    const Feat feats[6] = {
        { f4,   512, 38, 38,    0 },
        { f7,  1024, 19, 19, 5776 },
        { f8,   512, 10, 10, 7942 },
        { f9,   256,  5,  5, 8542 },
        { f10,  256,  3,  3, 8692 },
        { f11,  256,  1,  1, 8728 },
    };
    const int locPi[6] = {23, 24, 25, 26, 27, 28};
    const int clsPi[6] = {29, 30, 31, 32, 33, 34};
    const int nbox[6]  = {4, 6, 6, 6, 4, 4};

    for (int i = 0; i < 6; ++i) {
        const Feat& F = feats[i];
        int Co = nbox[i] * 4;   // loc head -> [N, TB, 4] region
        conv(F.x, locPi[i], locs, 1, F.C, F.C, F.H, F.W, Co, F.H, F.W,
             3, 1, 1, 1, 0, (long)TB * 4, (long)F.boxOff * 4);
        Co = nbox[i] * 2;       // cls head -> [N, TB, 2] region
        conv(F.x, clsPi[i], scrs, 1, F.C, F.C, F.H, F.W, Co, F.H, F.W,
             3, 1, 1, 1, 0, (long)TB * 2, (long)F.boxOff * 2);
    }
}
